// HGN3View_MoE_86371792322713
// MI455X (gfx1250) — compile-verified, matrix operand order checked by reading, not run on hardware
//
#include <hip/hip_runtime.h>
#include <hip/hip_bf16.h>

// ---------------------------------------------------------------------------
// Problem constants (from the reference)
// ---------------------------------------------------------------------------
#define NN    40000
#define EE    640000
#define BB    8192
#define IN_   768
#define META_ 6
#define INM   774          // IN + META
#define INMP  800          // padded to mult of 32
#define HID_  256
#define OUT_  128
#define NE_   4
#define REL_  200
#define NET_  3
#define NL_   4
#define NH_   4
#define FF_   512
#define SEQ_  3
#define TOK_  (BB*SEQ_)    // 24576

typedef _Float16 v16h __attribute__((ext_vector_type(16)));
typedef _Float16 v8h  __attribute__((ext_vector_type(8)));
typedef float    v8f  __attribute__((ext_vector_type(8)));

union FragH { v16h v; v8h h[2]; };

// ---------------------------------------------------------------------------
// WMMA GEMM:  C[M x N] = act( A[M x K] @ W[N x K]^T + bias )
// A, W are f16 row-major with row strides lda/ldw (in elements).
// Kp must be a multiple of 32; M a multiple of 32; N a multiple of 64.
// Block = 256 threads = 8 waves, arranged 4(M) x 2(N); wave tile = 32x32.
// Block tile = 128 x 64.
// ---------------------------------------------------------------------------
__global__ void wmma_gemm(const _Float16* __restrict__ A, int lda,
                          const _Float16* __restrict__ W, int ldw,
                          const float* __restrict__ bias,
                          float* __restrict__ C, _Float16* __restrict__ Ch,
                          int M, int N, int Kp, int act)
{
  const int lane = threadIdx.x & 31;
  const int wave = threadIdx.x >> 5;
  const int wm = wave & 3, wn = wave >> 2;
  const int m0 = blockIdx.x * 128 + wm * 32;
  const int n0 = blockIdx.y * 64 + wn * 32;
  if (m0 >= M) return;                     // wave-uniform (M % 32 == 0)

  const int lm   = lane & 15;
  const int koff = (lane >> 4) * 8;        // lanes 16-31 start at K=+8

  v8f acc00 = {}, acc01 = {}, acc10 = {}, acc11 = {};

  const _Float16* Ar0 = A + (size_t)(m0 + lm)      * lda;
  const _Float16* Ar1 = A + (size_t)(m0 + 16 + lm) * lda;
  const _Float16* Wr0 = W + (size_t)(n0 + lm)      * ldw;
  const _Float16* Wr1 = W + (size_t)(n0 + 16 + lm) * ldw;

  for (int kk = 0; kk < Kp; kk += 32) {
    const int kb = kk + koff;
    FragH a0, a1, b0, b1;
    a0.h[0] = *(const v8h*)(Ar0 + kb);
    a0.h[1] = *(const v8h*)(Ar0 + kb + 16);
    a1.h[0] = *(const v8h*)(Ar1 + kb);
    a1.h[1] = *(const v8h*)(Ar1 + kb + 16);
    b0.h[0] = *(const v8h*)(Wr0 + kb);
    b0.h[1] = *(const v8h*)(Wr0 + kb + 16);
    b1.h[0] = *(const v8h*)(Wr1 + kb);
    b1.h[1] = *(const v8h*)(Wr1 + kb + 16);
    // prefetch next K block (lowers to global_prefetch_b8)
    __builtin_prefetch((const void*)(Ar0 + kb + 32), 0, 3);
    __builtin_prefetch((const void*)(Wr0 + kb + 32), 0, 3);

    acc00 = __builtin_amdgcn_wmma_f32_16x16x32_f16(false, a0.v, false, b0.v, (short)0, acc00, false, false);
    acc01 = __builtin_amdgcn_wmma_f32_16x16x32_f16(false, a0.v, false, b1.v, (short)0, acc01, false, false);
    acc10 = __builtin_amdgcn_wmma_f32_16x16x32_f16(false, a1.v, false, b0.v, (short)0, acc10, false, false);
    acc11 = __builtin_amdgcn_wmma_f32_16x16x32_f16(false, a1.v, false, b1.v, (short)0, acc11, false, false);
  }

  // C/D layout: VGPR r -> lanes 0-15: (M=r, N=lane); lanes 16-31: (M=r+8, N=lane-16)
  const int rowOff = (lane >> 4) * 8;
  v8f* accp[2][2] = { { &acc00, &acc01 }, { &acc10, &acc11 } };
  for (int t = 0; t < 2; ++t) {
    const int rowT = m0 + t * 16 + rowOff;
    for (int u = 0; u < 2; ++u) {
      const int col = n0 + u * 16 + lm;
      const float bv = bias ? bias[col] : 0.0f;
      v8f a = *accp[t][u];
#pragma unroll
      for (int r = 0; r < 8; ++r) {
        float v = a[r] + bv;
        if (act == 1) v = v > 0.f ? v : 0.f;
        const size_t idx = (size_t)(rowT + r) * N + col;
        if (C)  C[idx]  = v;
        if (Ch) Ch[idx] = (_Float16)v;
      }
    }
  }
}

// ---------------------------------------------------------------------------
// Elementwise / utility kernels
// ---------------------------------------------------------------------------
__global__ void k_fill(float* __restrict__ p, float v, long long n) {
  long long i = (long long)blockIdx.x * blockDim.x + threadIdx.x;
  if (i < n) p[i] = v;
}

// f32 -> f16 with K padding (zero-fill k >= K)
__global__ void k_cvt(const float* __restrict__ s, _Float16* __restrict__ d,
                      long long total, int K, int Kp) {
  long long i = (long long)blockIdx.x * blockDim.x + threadIdx.x;
  if (i >= total) return;
  long long r = i / Kp; int k = (int)(i % Kp);
  d[i] = (k < K) ? (_Float16)s[r * K + k] : (_Float16)0.0f;
}

// meta encoder stage 1: xm1h[b][n] = relu(meta[b] . me_w1[n] + b1[n]), K=6
__global__ void k_meta1(const float* __restrict__ x, const float* __restrict__ w1,
                        const float* __restrict__ b1, _Float16* __restrict__ out) {
  int i = blockIdx.x * blockDim.x + threadIdx.x;
  if (i >= BB * HID_) return;
  int b = i / HID_, n = i % HID_;
  float s = b1[n];
  const float* mr = x + (long long)b * INM + IN_;
  for (int j = 0; j < META_; ++j) s += mr[j] * w1[n * META_ + j];
  out[i] = (_Float16)(s > 0.f ? s : 0.f);
}

// per-node attention scores: s_i = Wx . a[0:256], s_j = Wx . a[256:512]
__global__ void k_att_scores(const float* __restrict__ Wx, const float* __restrict__ a,
                             float* __restrict__ s_i, float* __restrict__ s_j) {
  int i = blockIdx.x * blockDim.x + threadIdx.x;
  if (i >= NN) return;
  float si = 0.f, sj = 0.f;
  const float* w = Wx + (long long)i * HID_;
  for (int c = 0; c < HID_; ++c) { float v = w[c]; si += v * a[c]; sj += v * a[HID_ + c]; }
  s_i[i] = si; s_j[i] = sj;
}

// r_s[t] = rel[t] . (Wr^T @ a[512:768]);  Wr is (HID x REL)
__global__ void k_rel_scores(const float* __restrict__ Wr, const float* __restrict__ a,
                             const float* __restrict__ rel, float* __restrict__ r_s) {
  __shared__ float u[REL_];
  int t = threadIdx.x;
  if (t < REL_) {
    float s = 0.f;
    for (int d = 0; d < HID_; ++d) s += Wr[d * REL_ + t] * a[2 * HID_ + d];
    u[t] = s;
  }
  __syncthreads();
  if (t < NET_) {
    float s = 0.f;
    for (int r = 0; r < REL_; ++r) s += rel[t * REL_ + r] * u[r];
    r_s[t] = s;
  }
}

__device__ __forceinline__ void atomicMaxF(float* addr, float val) {
  if (val >= 0.f) atomicMax((int*)addr, __float_as_int(val));
  else            atomicMin((unsigned int*)addr, __float_as_uint(val));
}

__global__ void k_edge_logit(const int* __restrict__ src, const int* __restrict__ dst,
                             const int* __restrict__ et, const float* __restrict__ s_i,
                             const float* __restrict__ s_j, const float* __restrict__ r_s,
                             float* __restrict__ logit, float* __restrict__ m) {
  int e = blockIdx.x * blockDim.x + threadIdx.x;
  if (e >= EE) return;
  float v = s_i[dst[e]] + s_j[src[e]] + r_s[et[e]];
  v = v > 0.f ? v : 0.2f * v;          // leaky_relu 0.2
  logit[e] = v;
  atomicMaxF(&m[dst[e]], v);
}

__global__ void k_edge_exp(const int* __restrict__ dst, float* __restrict__ logit,
                           const float* __restrict__ m, float* __restrict__ ssum) {
  int e = blockIdx.x * blockDim.x + threadIdx.x;
  if (e >= EE) return;
  float ev = __expf(logit[e] - m[dst[e]]);
  logit[e] = ev;
  atomicAdd(&ssum[dst[e]], ev);
}

// one block (256 thr) per edge: agg[dst] += Wx[src] * alpha
__global__ void k_edge_agg(const int* __restrict__ src, const int* __restrict__ dst,
                           const float* __restrict__ e_, const float* __restrict__ ssum,
                           const float* __restrict__ Wx, float* __restrict__ agg) {
  int e = blockIdx.x, c = threadIdx.x;
  int d = dst[e], s = src[e];
  float alpha = e_[e] / (ssum[d] + 1e-16f);
  atomicAdd(&agg[(long long)d * HID_ + c], Wx[(long long)s * HID_ + c] * alpha);
}

__global__ void k_elu_out(const float* __restrict__ agg, const float* __restrict__ res,
                          float* __restrict__ h, _Float16* __restrict__ hh, long long n) {
  long long i = (long long)blockIdx.x * blockDim.x + threadIdx.x;
  if (i >= n) return;
  float v = agg[i] + res[i];
  v = v > 0.f ? v : (__expf(v) - 1.f);   // elu
  h[i] = v;
  if (hh) hh[i] = (_Float16)v;
}

__global__ void k_rownorm(float* __restrict__ h, _Float16* __restrict__ hh) {
  int i = blockIdx.x * blockDim.x + threadIdx.x;
  if (i >= NN) return;
  float s = 0.f;
  long long o = (long long)i * HID_;
  for (int c = 0; c < HID_; ++c) { float v = h[o + c]; s += v * v; }
  float inv = 1.f / fmaxf(sqrtf(s), 1e-12f);
  for (int c = 0; c < HID_; ++c) { float v = h[o + c] * inv; h[o + c] = v; hh[o + c] = (_Float16)v; }
}

// MoE gating: noisy logits, drop bottom NE-K, softmax -> combine weights
__global__ void k_moe_gate(const float* __restrict__ xv, const float* __restrict__ gate_w,
                           const float* __restrict__ noise_w, const float* __restrict__ eps,
                           float* __restrict__ comb, float* __restrict__ Lsum) {
  int b = blockIdx.x * blockDim.x + threadIdx.x;
  if (b >= BB) return;
  const float* xr = xv + (long long)b * OUT_;
  float g[NE_];
  for (int e = 0; e < NE_; ++e) {
    float s = 0.f;
    for (int c = 0; c < OUT_; ++c) s += xr[c] * gate_w[e * OUT_ + c];
    g[e] = s;
  }
  float nv = 0.f;
  for (int c = 0; c < OUT_; ++c) nv += xr[c] * noise_w[c];
  float nsd = nv > 0.f ? nv + log1pf(__expf(-nv)) : log1pf(__expf(nv));  // softplus
  float h[NE_];
  for (int e = 0; e < NE_; ++e) h[e] = g[e] + eps[b * NE_ + e] * nsd;
  // drop two smallest
  int i0 = 0;
  for (int e = 1; e < NE_; ++e) if (h[e] < h[i0]) i0 = e;
  int i1 = -1;
  for (int e = 0; e < NE_; ++e) if (e != i0 && (i1 < 0 || h[e] < h[i1])) i1 = e;
  float hm[NE_];
  for (int e = 0; e < NE_; ++e) hm[e] = (e == i0 || e == i1) ? -3.4e38f : h[e];
  float mx = hm[0];
  for (int e = 1; e < NE_; ++e) mx = fmaxf(mx, hm[e]);
  float L[NE_], Z = 0.f;
  for (int e = 0; e < NE_; ++e) { L[e] = __expf(hm[e] - mx); Z += L[e]; }
  for (int e = 0; e < NE_; ++e) {
    float v = L[e] / Z;
    comb[b * NE_ + e] = v;
    atomicAdd(&Lsum[e], v);
  }
}

__global__ void k_moe_combine(const float* __restrict__ comb, const float* __restrict__ eo,
                              float* __restrict__ z, int view) {
  int i = blockIdx.x * blockDim.x + threadIdx.x;
  if (i >= BB * OUT_) return;
  int b = i / OUT_, o = i % OUT_;
  float s = 0.f;
  for (int e = 0; e < NE_; ++e) s += comb[b * NE_ + e] * eo[((long long)e * BB + b) * OUT_ + o];
  z[(long long)b * (SEQ_ * OUT_) + view * OUT_ + o] = s;
}

// S=3 multi-head attention: one thread per (batch, head)
__global__ void k_attn(const float* __restrict__ qkv, float* __restrict__ o) {
  int t = blockIdx.x * blockDim.x + threadIdx.x;
  if (t >= BB * NH_) return;
  int b = t >> 2, hd = t & 3;
  const int HD = OUT_ / NH_;  // 32
  const float scale = 0.17677669529663687f;  // 1/sqrt(32)
  const float* base = qkv + (long long)b * SEQ_ * (3 * OUT_);
  float att[SEQ_][SEQ_];
  for (int s = 0; s < SEQ_; ++s) {
    const float* q = base + s * (3 * OUT_) + hd * HD;
    for (int u = 0; u < SEQ_; ++u) {
      const float* k = base + u * (3 * OUT_) + OUT_ + hd * HD;
      float d = 0.f;
      for (int c = 0; c < HD; ++c) d += q[c] * k[c];
      att[s][u] = d * scale;
    }
  }
  for (int s = 0; s < SEQ_; ++s) {
    float mx = fmaxf(att[s][0], fmaxf(att[s][1], att[s][2]));
    float Z = 0.f;
    for (int u = 0; u < SEQ_; ++u) { att[s][u] = __expf(att[s][u] - mx); Z += att[s][u]; }
    for (int u = 0; u < SEQ_; ++u) att[s][u] /= Z;
  }
  for (int s = 0; s < SEQ_; ++s) {
    float* orow = o + ((long long)b * SEQ_ + s) * OUT_ + hd * HD;
    for (int c = 0; c < HD; ++c) {
      float acc = 0.f;
      for (int u = 0; u < SEQ_; ++u)
        acc += att[s][u] * base[u * (3 * OUT_) + 2 * OUT_ + hd * HD + c];
      orow[c] = acc;
    }
  }
}

// LayerNorm( x + y ) with scale/shift; writes f32 + f16
__global__ void k_add_ln(const float* __restrict__ x, const float* __restrict__ y,
                         const float* __restrict__ g, const float* __restrict__ b,
                         float* __restrict__ out, _Float16* __restrict__ outh, int rows) {
  int r = blockIdx.x * blockDim.x + threadIdx.x;
  if (r >= rows) return;
  long long o = (long long)r * OUT_;
  float mu = 0.f;
  for (int c = 0; c < OUT_; ++c) mu += x[o + c] + y[o + c];
  mu *= (1.f / OUT_);
  float var = 0.f;
  for (int c = 0; c < OUT_; ++c) { float d = x[o + c] + y[o + c] - mu; var += d * d; }
  var *= (1.f / OUT_);
  float inv = rsqrtf(var + 1e-5f);
  for (int c = 0; c < OUT_; ++c) {
    float v = (x[o + c] + y[o + c] - mu) * inv * g[c] + b[c];
    out[o + c] = v; outh[o + c] = (_Float16)v;
  }
}

__global__ void k_clf(const float* __restrict__ z, const float* __restrict__ w,
                      const float* __restrict__ b, float* __restrict__ out) {
  int i = blockIdx.x * blockDim.x + threadIdx.x;
  if (i >= BB * 2) return;
  int bb = i / 2, c = i % 2;
  float s = b[c];
  const float* zr = z + (long long)bb * (SEQ_ * OUT_);
  for (int k = 0; k < SEQ_ * OUT_; ++k) s += zr[k] * w[c * (SEQ_ * OUT_) + k];
  out[(long long)bb * 2 + c] = s;
}

__global__ void k_loss(const float* __restrict__ Lsum, float* __restrict__ out) {
  if (threadIdx.x != 0 || blockIdx.x != 0) return;
  float total = 0.f;
  for (int v = 0; v < 3; ++v) {
    const float* t = Lsum + v * NE_;
    float mu = 0.f;
    for (int e = 0; e < NE_; ++e) mu += t[e];
    mu *= (1.f / NE_);
    float var = 0.f;
    for (int e = 0; e < NE_; ++e) { float d = t[e] - mu; var += d * d; }
    var *= (1.f / (NE_ - 1));          // ddof=1
    total += var / (mu * mu);
  }
  out[(long long)BB * 2] = total;
}

// ---------------------------------------------------------------------------
// Host driver
// ---------------------------------------------------------------------------
extern "C" void kernel_launch(void* const* d_in, const int* in_sizes, int n_in,
                              void* d_out, int out_size, void* d_ws, size_t ws_size,
                              hipStream_t stream) {
  (void)in_sizes; (void)n_in; (void)out_size; (void)ws_size;
  // ---- inputs (setup_inputs dict order) ----
  const float* x         = (const float*)d_in[0];
  const float* noise     = (const float*)d_in[1];
  const float* g_lin1_w  = (const float*)d_in[2];
  const float* g_lin1_b  = (const float*)d_in[3];
  const float* conv_W    = (const float*)d_in[4];
  const float* conv_Wr   = (const float*)d_in[5];
  const float* conv_a    = (const float*)d_in[6];
  const float* conv_Wres = (const float*)d_in[7];
  const float* conv_rel  = (const float*)d_in[8];
  const float* g_lin2_w  = (const float*)d_in[9];
  const float* g_lin2_b  = (const float*)d_in[10];
  const float* te_w      = (const float*)d_in[11];
  const float* te_b      = (const float*)d_in[12];
  const float* me_w1     = (const float*)d_in[13];
  const float* me_b1     = (const float*)d_in[14];
  const float* me_w2     = (const float*)d_in[15];
  const float* me_b2     = (const float*)d_in[16];
  const float* moe_gate_w  = (const float*)d_in[17];
  const float* moe_noise_w = (const float*)d_in[18];
  const float* moe_w1    = (const float*)d_in[19];
  const float* moe_b1    = (const float*)d_in[20];
  const float* moe_w2    = (const float*)d_in[21];
  const float* moe_b2    = (const float*)d_in[22];
  const float* tf_qkv_w  = (const float*)d_in[23];
  const float* tf_qkv_b  = (const float*)d_in[24];
  const float* tf_out_w  = (const float*)d_in[25];
  const float* tf_out_b  = (const float*)d_in[26];
  const float* tf_ln1_g  = (const float*)d_in[27];
  const float* tf_ln1_b  = (const float*)d_in[28];
  const float* tf_ff1_w  = (const float*)d_in[29];
  const float* tf_ff1_b  = (const float*)d_in[30];
  const float* tf_ff2_w  = (const float*)d_in[31];
  const float* tf_ff2_b  = (const float*)d_in[32];
  const float* tf_ln2_g  = (const float*)d_in[33];
  const float* tf_ln2_b  = (const float*)d_in[34];
  const float* clf_w     = (const float*)d_in[35];
  const float* clf_b     = (const float*)d_in[36];
  const int*   edge_index = (const int*)d_in[37];
  const int*   edge_type  = (const int*)d_in[38];
  float* out = (float*)d_out;

  const int* e_src = edge_index;
  const int* e_dst = edge_index + EE;

  // ---- workspace arena ----
  char* ws = (char*)d_ws;
  size_t off = 0;
  auto alloc = [&](size_t bytes) -> void* {
    void* p = ws + off;
    off = (off + bytes + 255) & ~(size_t)255;
    return p;
  };
  _Float16* xh       = (_Float16*)alloc((size_t)NN * INMP * 2);
  _Float16* w1h      = (_Float16*)alloc((size_t)HID_ * INMP * 2);
  float*    hf       = (float*)   alloc((size_t)NN * HID_ * 4);
  _Float16* hh       = (_Float16*)alloc((size_t)NN * HID_ * 2);
  _Float16* convWh   = (_Float16*)alloc((size_t)2 * HID_ * HID_ * 2);
  _Float16* convWresh= (_Float16*)alloc((size_t)2 * HID_ * HID_ * 2);
  float*    Wx       = (float*)   alloc((size_t)NN * HID_ * 4);
  float*    res      = (float*)   alloc((size_t)NN * HID_ * 4);
  float*    agg      = (float*)   alloc((size_t)NN * HID_ * 4);
  float*    s_i      = (float*)   alloc((size_t)NN * 4);
  float*    s_j      = (float*)   alloc((size_t)NN * 4);
  float*    mmax     = (float*)   alloc((size_t)NN * 4);
  float*    ssum     = (float*)   alloc((size_t)NN * 4);
  float*    r_s      = (float*)   alloc(256);
  float*    elog     = (float*)   alloc((size_t)EE * 4);
  _Float16* glin2h   = (_Float16*)alloc((size_t)OUT_ * HID_ * 2);
  float*    graphf   = (float*)   alloc((size_t)NN * OUT_ * 4);
  _Float16* graphh   = (_Float16*)alloc((size_t)NN * OUT_ * 2);
  _Float16* tewh     = (_Float16*)alloc((size_t)OUT_ * IN_ * 2);
  float*    xt_f     = (float*)   alloc((size_t)BB * OUT_ * 4);
  _Float16* xt_h     = (_Float16*)alloc((size_t)BB * OUT_ * 2);
  _Float16* xm1h     = (_Float16*)alloc((size_t)BB * HID_ * 2);
  _Float16* mew2h    = (_Float16*)alloc((size_t)OUT_ * HID_ * 2);
  float*    xm_f     = (float*)   alloc((size_t)BB * OUT_ * 4);
  _Float16* xm_h     = (_Float16*)alloc((size_t)BB * OUT_ * 2);
  _Float16* moew1h   = (_Float16*)alloc((size_t)3 * NE_ * (4*OUT_) * OUT_ * 2);
  _Float16* moew2h   = (_Float16*)alloc((size_t)3 * NE_ * OUT_ * (4*OUT_) * 2);
  float*    comb     = (float*)   alloc((size_t)BB * NE_ * 4);
  float*    Lsum     = (float*)   alloc(3 * NE_ * 4);
  _Float16* h1h      = (_Float16*)alloc((size_t)BB * FF_ * 2);
  float*    eo       = (float*)   alloc((size_t)NE_ * BB * OUT_ * 4);
  float*    z        = (float*)   alloc((size_t)TOK_ * OUT_ * 4);
  _Float16* zh       = (_Float16*)alloc((size_t)TOK_ * OUT_ * 2);
  _Float16* tfqkvh   = (_Float16*)alloc((size_t)NL_ * (3*OUT_) * OUT_ * 2);
  _Float16* tfouth   = (_Float16*)alloc((size_t)NL_ * OUT_ * OUT_ * 2);
  _Float16* tff1h    = (_Float16*)alloc((size_t)NL_ * FF_ * OUT_ * 2);
  _Float16* tff2h    = (_Float16*)alloc((size_t)NL_ * OUT_ * FF_ * 2);
  float*    qkvb     = (float*)   alloc((size_t)TOK_ * (3*OUT_) * 4);
  float*    ob       = (float*)   alloc((size_t)TOK_ * OUT_ * 4);
  _Float16* oh       = (_Float16*)alloc((size_t)TOK_ * OUT_ * 2);
  float*    projb    = (float*)   alloc((size_t)TOK_ * OUT_ * 4);
  _Float16* ff1h     = (_Float16*)alloc((size_t)TOK_ * FF_ * 2);
  float*    ff2b     = (float*)   alloc((size_t)TOK_ * OUT_ * 4);

  // ---- launch helpers ----
  auto gemm = [&](const _Float16* A, int lda, const _Float16* Wp, int ldw,
                  const float* bias, float* C, _Float16* Ch,
                  int M, int N, int Kp, int act) {
    dim3 g((unsigned)((M + 127) / 128), (unsigned)(N / 64));
    wmma_gemm<<<g, 256, 0, stream>>>(A, lda, Wp, ldw, bias, C, Ch, M, N, Kp, act);
  };
  auto cvt = [&](const float* s, _Float16* d, long long rows, int K, int Kp) {
    long long tot = rows * Kp;
    k_cvt<<<(unsigned)((tot + 255) / 256), 256, 0, stream>>>(s, d, tot, K, Kp);
  };
  auto fill = [&](float* p, float v, long long n) {
    k_fill<<<(unsigned)((n + 255) / 256), 256, 0, stream>>>(p, v, n);
  };

  // ================= stage 1: input GEMM ====================
  cvt(x, xh, NN, INM, INMP);
  cvt(g_lin1_w, w1h, HID_, INM, INMP);
  gemm(xh, INMP, w1h, INMP, g_lin1_b, hf, hh, NN, HID_, INMP, /*relu*/1);

  // ================= stage 2: two HGN layers ================
  cvt(conv_W,    convWh,    2 * HID_, HID_, HID_);
  cvt(conv_Wres, convWresh, 2 * HID_, HID_, HID_);
  for (int l = 0; l < 2; ++l) {
    gemm(hh, HID_, convWh    + (size_t)l * HID_ * HID_, HID_, nullptr, Wx,  nullptr, NN, HID_, HID_, 0);
    gemm(hh, HID_, convWresh + (size_t)l * HID_ * HID_, HID_, nullptr, res, nullptr, NN, HID_, HID_, 0);
    k_att_scores<<<(NN + 255) / 256, 256, 0, stream>>>(Wx, conv_a + (size_t)l * 3 * HID_, s_i, s_j);
    k_rel_scores<<<1, 256, 0, stream>>>(conv_Wr + (size_t)l * HID_ * REL_,
                                        conv_a + (size_t)l * 3 * HID_,
                                        conv_rel + (size_t)l * NET_ * REL_, r_s);
    fill(mmax, -3.4e38f, NN);
    fill(ssum, 0.f, NN);
    fill(agg,  0.f, (long long)NN * HID_);
    k_edge_logit<<<(EE + 255) / 256, 256, 0, stream>>>(e_src, e_dst, edge_type, s_i, s_j, r_s, elog, mmax);
    k_edge_exp<<<(EE + 255) / 256, 256, 0, stream>>>(e_dst, elog, mmax, ssum);
    k_edge_agg<<<EE, HID_, 0, stream>>>(e_src, e_dst, elog, ssum, Wx, agg);
    k_elu_out<<<(unsigned)(((long long)NN * HID_ + 255) / 256), 256, 0, stream>>>(agg, res, hf, hh, (long long)NN * HID_);
    if (l == 1) k_rownorm<<<(NN + 255) / 256, 256, 0, stream>>>(hf, hh);
  }

  // ================= stage 3: three views ===================
  cvt(g_lin2_w, glin2h, OUT_, HID_, HID_);
  gemm(hh, HID_, glin2h, HID_, g_lin2_b, graphf, graphh, NN, OUT_, HID_, 1);

  cvt(te_w, tewh, OUT_, IN_, IN_);
  gemm(xh, INMP, tewh, IN_, te_b, xt_f, xt_h, BB, OUT_, IN_, 1);   // text = x[:B,:768]

  k_meta1<<<(BB * HID_ + 255) / 256, 256, 0, stream>>>(x, me_w1, me_b1, xm1h);
  cvt(me_w2, mew2h, OUT_, HID_, HID_);
  gemm(xm1h, HID_, mew2h, HID_, me_b2, xm_f, xm_h, BB, OUT_, HID_, 1);

  // ================= stage 4: MoE ===========================
  cvt(moe_w1, moew1h, (long long)3 * NE_ * (4 * OUT_), OUT_, OUT_);
  cvt(moe_w2, moew2h, (long long)3 * NE_ * OUT_, 4 * OUT_, 4 * OUT_);
  fill(Lsum, 0.f, 3 * NE_);
  const float*    xv_f[3] = { graphf, xt_f, xm_f };
  const _Float16* xv_h[3] = { graphh, xt_h, xm_h };
  for (int v = 0; v < 3; ++v) {
    k_moe_gate<<<(BB + 255) / 256, 256, 0, stream>>>(
        xv_f[v], moe_gate_w + (size_t)v * NE_ * OUT_, moe_noise_w + (size_t)v * OUT_,
        noise + (size_t)v * BB * NE_, comb, Lsum + v * NE_);
    for (int e = 0; e < NE_; ++e) {
      const size_t wi = ((size_t)v * NE_ + e);
      gemm(xv_h[v], OUT_, moew1h + wi * (4 * OUT_) * OUT_, OUT_,
           moe_b1 + wi * (4 * OUT_), nullptr, h1h, BB, FF_, OUT_, 1);
      gemm(h1h, FF_, moew2h + wi * OUT_ * (4 * OUT_), 4 * OUT_,
           moe_b2 + wi * OUT_, eo + (size_t)e * BB * OUT_, nullptr, BB, OUT_, FF_, 0);
    }
    k_moe_combine<<<(BB * OUT_ + 255) / 256, 256, 0, stream>>>(comb, eo, z, v);
  }

  // ================= stage 5: transformer ===================
  cvt(tf_qkv_w, tfqkvh, (long long)NL_ * 3 * OUT_, OUT_, OUT_);
  cvt(tf_out_w, tfouth, (long long)NL_ * OUT_, OUT_, OUT_);
  cvt(tf_ff1_w, tff1h,  (long long)NL_ * FF_, OUT_, OUT_);
  cvt(tf_ff2_w, tff2h,  (long long)NL_ * OUT_, FF_, FF_);
  for (int l = 0; l < NL_; ++l) {
    cvt(z, zh, TOK_, OUT_, OUT_);
    gemm(zh, OUT_, tfqkvh + (size_t)l * 3 * OUT_ * OUT_, OUT_,
         tf_qkv_b + (size_t)l * 3 * OUT_, qkvb, nullptr, TOK_, 3 * OUT_, OUT_, 0);
    k_attn<<<(BB * NH_ + 255) / 256, 256, 0, stream>>>(qkvb, ob);
    cvt(ob, oh, TOK_, OUT_, OUT_);
    gemm(oh, OUT_, tfouth + (size_t)l * OUT_ * OUT_, OUT_,
         tf_out_b + (size_t)l * OUT_, projb, nullptr, TOK_, OUT_, OUT_, 0);
    k_add_ln<<<(TOK_ + 255) / 256, 256, 0, stream>>>(
        z, projb, tf_ln1_g + (size_t)l * OUT_, tf_ln1_b + (size_t)l * OUT_, z, zh, TOK_);
    gemm(zh, OUT_, tff1h + (size_t)l * FF_ * OUT_, OUT_,
         tf_ff1_b + (size_t)l * FF_, nullptr, ff1h, TOK_, FF_, OUT_, 1);
    gemm(ff1h, FF_, tff2h + (size_t)l * OUT_ * FF_, FF_,
         tf_ff2_b + (size_t)l * OUT_, ff2b, nullptr, TOK_, OUT_, FF_, 0);
    k_add_ln<<<(TOK_ + 255) / 256, 256, 0, stream>>>(
        z, ff2b, tf_ln2_g + (size_t)l * OUT_, tf_ln2_b + (size_t)l * OUT_, z, zh, TOK_);
  }

  // ================= stage 6: classifier + loss =============
  k_clf<<<(BB * 2 + 255) / 256, 256, 0, stream>>>(z, clf_w, clf_b, out);
  k_loss<<<1, 1, 0, stream>>>(Lsum, out);
}
